// BaseLayer_70128226009754
// MI455X (gfx1250) — matrix operand order repaired and checked by reading, not compiled, not run on hardware
//
#include <hip/hip_runtime.h>
#include <hip/hip_bf16.h>
#include <stdint.h>

// ---------------------------------------------------------------------------
// Problem constants (from reference: B=4, S=2048, D=1024, F=4096, E=8)
// ---------------------------------------------------------------------------
#define T_TOK 8192
#define DIM   1024
#define FFN   4096
#define NEXP  8
#define CAP   1024   // T/E

typedef __bf16 bf16_t;
typedef bf16_t v16bf __attribute__((ext_vector_type(16)));
typedef bf16_t v8bf  __attribute__((ext_vector_type(8)));
typedef float  v8f   __attribute__((ext_vector_type(8)));

// f32 -> bf16 round-to-nearest-even
static __device__ __forceinline__ bf16_t f2bf(float f) {
  union { float f; uint32_t u; } v; v.f = f;
  uint32_t r = v.u + 0x7FFFu + ((v.u >> 16) & 1u);
  uint16_t h = (uint16_t)(r >> 16);
  bf16_t out;
  __builtin_memcpy(&out, &h, 2);
  return out;
}

// branch-free tanh-approx GELU: tanh(y) via v_exp_f32 + v_rcp_f32
static __device__ __forceinline__ float gelu_fast(float x) {
  const float k0 = 0.7978845608028654f;   // sqrt(2/pi)
  const float k1 = 0.044715f;
  float y = k0 * (x + k1 * x * x * x);
  float t = __expf(-2.0f * fabsf(y));                    // v_exp_f32
  float th = (1.0f - t) * __builtin_amdgcn_rcpf(1.0f + t); // v_rcp_f32
  th = copysignf(th, y);
  return 0.5f * x * (1.0f + th);
}

static __device__ __forceinline__ float wave_sum(float v) {
  #pragma unroll
  for (int off = 16; off > 0; off >>= 1)
    v += __shfl_xor(v, off, 32);
  return v;
}

static __device__ __forceinline__ v8f wmma_bf16(v16bf a, v16bf b, v8f c) {
  return __builtin_amdgcn_wmma_f32_16x16x32_bf16(
      /*neg_a=*/false, a, /*neg_b=*/false, b,
      /*c_mod=*/(short)0, c, /*reuse_a=*/false, /*reuse_b=*/false);
}

// A fragment (16x32 bf16) from LDS tile Alds[128][32]:
// lane L: M = L%16 ; K-base = (L/16)*8 ; VGPR0..3 = K kb..kb+7, VGPR4..7 = K 16+kb..16+kb+7
static __device__ __forceinline__ v16bf load_a_frag(const bf16_t* Alds, int rowBase, int lane) {
  int m  = lane & 15;
  int kb = (lane >> 4) * 8;
  const bf16_t* rp = Alds + (rowBase + m) * 32;
  v8bf lo = *(const v8bf*)(rp + kb);
  v8bf hi = *(const v8bf*)(rp + 16 + kb);
  v16bf a;
  #pragma unroll
  for (int i = 0; i < 8; ++i) { a[i] = lo[i]; a[i + 8] = hi[i]; }
  return a;
}

// B fragment (32x16 bf16) from LDS tile Blds[32][128]:
// lane L: N = cn + L%16 ; K-base = (L/16)*8 ; elems 0..7 = K kb..kb+7, elems 8..15 = K 16+kb..16+kb+7
static __device__ __forceinline__ v16bf load_b_frag(const bf16_t* Blds, int cn, int lane) {
  const bf16_t* rp = Blds + cn + (lane & 15);
  int kb = (lane >> 4) * 8;
  v16bf b;
  #pragma unroll
  for (int i = 0; i < 8; ++i) { b[i] = rp[(kb + i) * 128]; b[i + 8] = rp[(16 + kb + i) * 128]; }
  return b;
}

// ---------------------------------------------------------------------------
// Kernel 1: per-token LayerNorm (f32 stats) -> norm bf16 [T,D], scores f32 [T,E]
// ---------------------------------------------------------------------------
__global__ __launch_bounds__(256) void ln_scores_kernel(
    const float* __restrict__ x, const float* __restrict__ cent,
    const float* __restrict__ g, const float* __restrict__ b,
    bf16_t* __restrict__ norm, float* __restrict__ scores) {
  int lane = threadIdx.x & 31;
  int wid  = threadIdx.x >> 5;
  int t = blockIdx.x * 8 + wid;
  const float* row = x + (size_t)t * DIM;

  float vals[32];
  float s = 0.f, ss = 0.f;
  #pragma unroll
  for (int i = 0; i < 32; ++i) {
    float v = row[lane + i * 32];
    vals[i] = v; s += v; ss += v * v;
  }
  s  = wave_sum(s);
  ss = wave_sum(ss);
  float mu   = s * (1.0f / DIM);
  float var  = ss * (1.0f / DIM) - mu * mu;
  float rstd = rsqrtf(var + 1e-5f);

  float dot[NEXP];
  #pragma unroll
  for (int e = 0; e < NEXP; ++e) dot[e] = 0.f;

  #pragma unroll
  for (int i = 0; i < 32; ++i) {
    int d = lane + i * 32;
    float nv = (vals[i] - mu) * rstd * g[d] + b[d];
    norm[(size_t)t * DIM + d] = f2bf(nv);
    #pragma unroll
    for (int e = 0; e < NEXP; ++e) dot[e] += nv * cent[e * DIM + d];
  }
  #pragma unroll
  for (int e = 0; e < NEXP; ++e) {
    float v = wave_sum(dot[e]);
    if (lane == 0) scores[(size_t)t * NEXP + e] = v;
  }
}

// ---------------------------------------------------------------------------
// Kernel 2: balanced assignment. Single block, 1024 threads.
// Greedy per-expert top-C via bitonic sort of 8192 (score,idx) keys in LDS.
// ---------------------------------------------------------------------------
__global__ __launch_bounds__(1024) void assign_kernel(
    const float* __restrict__ scores, int* __restrict__ sortOut,
    unsigned int* __restrict__ takenBits) {
  __shared__ uint64_t keys[T_TOK];   // 64 KB
  int tid = threadIdx.x;

  if (tid < T_TOK / 32) atomicExch(&takenBits[tid], 0u);
  __threadfence();
  __syncthreads();

  for (int e = 0; e < NEXP; ++e) {
    #pragma unroll
    for (int r = 0; r < 8; ++r) {
      int t = tid + r * 1024;
      unsigned int tk = (atomicOr(&takenBits[t >> 5], 0u) >> (t & 31)) & 1u;
      float sc = tk ? -__builtin_inff() : scores[(size_t)t * NEXP + e];
      uint32_t u = __float_as_uint(sc);
      u = (u & 0x80000000u) ? ~u : (u | 0x80000000u);
      keys[t] = ((uint64_t)u << 32) | (uint64_t)(8191 - t);
    }
    __syncthreads();

    for (uint32_t k2 = 2; k2 <= T_TOK; k2 <<= 1) {
      for (uint32_t j = k2 >> 1; j > 0; j >>= 1) {
        #pragma unroll
        for (int r = 0; r < 8; ++r) {
          uint32_t i   = (uint32_t)tid + r * 1024u;
          uint32_t ixj = i ^ j;
          if (ixj > i) {
            bool desc = ((i & k2) == 0);
            uint64_t a = keys[i], bb = keys[ixj];
            bool sw = desc ? (a < bb) : (a > bb);
            if (sw) { keys[i] = bb; keys[ixj] = a; }
          }
        }
        __syncthreads();
      }
    }

    if (tid < CAP) {
      int idx = 8191 - (int)(keys[tid] & 0x1FFFu);
      sortOut[e * CAP + tid] = idx;
      atomicOr(&takenBits[idx >> 5], 1u << (idx & 31));
    }
    __threadfence();
    __syncthreads();
  }
}

// ---------------------------------------------------------------------------
// Kernel 3: GEMM1  H1[e] = gelu( gather(norm, sort)[e] (CxD) @ w1[e] (DxF) + b1[e] )
// 128x128 tile, 8 waves (2x4), wave = 64x32 (4x2 WMMA tiles), K-step 32,
// double-buffered LDS with register staging to overlap HBM with WMMA.
// ---------------------------------------------------------------------------
__global__ __launch_bounds__(256) void gemm1_kernel(
    const bf16_t* __restrict__ norm, const float* __restrict__ w1,
    const float* __restrict__ b1, const int* __restrict__ sortIdx,
    bf16_t* __restrict__ H1) {
  const int e  = blockIdx.z;
  const int rb = blockIdx.y;   // 0..7   rows of C
  const int cb = blockIdx.x;   // 0..31  cols of F
  __shared__ bf16_t Alds[2][128 * 32];
  __shared__ bf16_t Blds[2][32 * 128];

  const int tid = threadIdx.x, lane = tid & 31, wid = tid >> 5;
  const int waveRow = (wid >> 2) * 64;
  const int waveCol = (wid & 3) * 32;

  v8f acc[4][2];
  #pragma unroll
  for (int m = 0; m < 4; ++m)
    #pragma unroll
    for (int n = 0; n < 2; ++n)
      #pragma unroll
      for (int i = 0; i < 8; ++i) acc[m][n][i] = 0.f;

  // A loader mapping: 2 threads/row, 16 bf16 each
  const int arow  = tid >> 1;
  const int ahalf = (tid & 1) * 16;
  const int srcRow = sortIdx[e * CAP + rb * 128 + arow];
  const bf16_t* aSrc = norm + (size_t)srcRow * DIM + ahalf;
  const int aDstOff = arow * 32 + ahalf;

  const float* wBase = w1 + (size_t)e * DIM * FFN + (size_t)cb * 128;

  uint4  aReg[2];
  float4 bReg[4];

  // prologue: tile 0 -> regs -> LDS buf 0
  {
    const uint4* s = (const uint4*)(aSrc);
    aReg[0] = s[0]; aReg[1] = s[1];
    #pragma unroll
    for (int i = 0; i < 4; ++i) {
      int fi = tid + i * 256, r = fi >> 5, c4 = (fi & 31) * 4;
      bReg[i] = *(const float4*)(wBase + (size_t)r * FFN + c4);
    }
    uint4* d = (uint4*)(Alds[0] + aDstOff);
    d[0] = aReg[0]; d[1] = aReg[1];
    #pragma unroll
    for (int i = 0; i < 4; ++i) {
      int fi = tid + i * 256, r = fi >> 5, c4 = (fi & 31) * 4;
      bf16_t* bd = Blds[0] + r * 128 + c4;
      bd[0] = f2bf(bReg[i].x); bd[1] = f2bf(bReg[i].y);
      bd[2] = f2bf(bReg[i].z); bd[3] = f2bf(bReg[i].w);
    }
  }
  __syncthreads();

  const int nk = DIM / 32;
  for (int kk = 0; kk < nk; ++kk) {
    const int cur = kk & 1, nxt = cur ^ 1;
    // issue global loads for tile kk+1 (latency hidden under WMMAs below)
    if (kk + 1 < nk) {
      int k0 = (kk + 1) * 32;
      const uint4* s = (const uint4*)(aSrc + k0);
      aReg[0] = s[0]; aReg[1] = s[1];
      #pragma unroll
      for (int i = 0; i < 4; ++i) {
        int fi = tid + i * 256, r = fi >> 5, c4 = (fi & 31) * 4;
        bReg[i] = *(const float4*)(wBase + (size_t)(k0 + r) * FFN + c4);
      }
    }
    // compute on buffer `cur`
    {
      v16bf bf0 = load_b_frag(Blds[cur], waveCol, lane);
      v16bf bf1 = load_b_frag(Blds[cur], waveCol + 16, lane);
      #pragma unroll
      for (int m = 0; m < 4; ++m) {
        v16bf af = load_a_frag(Alds[cur], waveRow + m * 16, lane);
        acc[m][0] = wmma_bf16(af, bf0, acc[m][0]);
        acc[m][1] = wmma_bf16(af, bf1, acc[m][1]);
      }
    }
    // drain staged registers into buffer `nxt`
    if (kk + 1 < nk) {
      uint4* d = (uint4*)(Alds[nxt] + aDstOff);
      d[0] = aReg[0]; d[1] = aReg[1];
      #pragma unroll
      for (int i = 0; i < 4; ++i) {
        int fi = tid + i * 256, r = fi >> 5, c4 = (fi & 31) * 4;
        bf16_t* bd = Blds[nxt] + r * 128 + c4;
        bd[0] = f2bf(bReg[i].x); bd[1] = f2bf(bReg[i].y);
        bd[2] = f2bf(bReg[i].z); bd[3] = f2bf(bReg[i].w);
      }
    }
    __syncthreads();
  }

  // epilogue: bias + gelu -> bf16 H1
  const int nLane = lane & 15;
  const int mOff  = (lane >> 4) * 8;
  #pragma unroll
  for (int m = 0; m < 4; ++m)
    #pragma unroll
    for (int n = 0; n < 2; ++n) {
      int col = cb * 128 + waveCol + n * 16 + nLane;
      float bias = b1[e * FFN + col];
      #pragma unroll
      for (int i = 0; i < 8; ++i) {
        int row = rb * 128 + waveRow + m * 16 + mOff + i;
        float v = gelu_fast(acc[m][n][i] + bias);
        H1[((size_t)e * CAP + row) * FFN + col] = f2bf(v);
      }
    }
}

// ---------------------------------------------------------------------------
// Kernel 4: GEMM2  out = routed + H1[e] (CxF) @ w2[e] (FxD) + b2[e],
// scattered to d_out[sort[row]] (fuses inverse permutation + residual)
// ---------------------------------------------------------------------------
__global__ __launch_bounds__(256) void gemm2_kernel(
    const bf16_t* __restrict__ H1, const float* __restrict__ w2,
    const float* __restrict__ b2, const int* __restrict__ sortIdx,
    const float* __restrict__ x, float* __restrict__ out) {
  const int e  = blockIdx.z;
  const int rb = blockIdx.y;   // 0..7 rows of C
  const int cb = blockIdx.x;   // 0..7 cols of D
  __shared__ bf16_t Alds[2][128 * 32];
  __shared__ bf16_t Blds[2][32 * 128];

  const int tid = threadIdx.x, lane = tid & 31, wid = tid >> 5;
  const int waveRow = (wid >> 2) * 64;
  const int waveCol = (wid & 3) * 32;

  v8f acc[4][2];
  #pragma unroll
  for (int m = 0; m < 4; ++m)
    #pragma unroll
    for (int n = 0; n < 2; ++n)
      #pragma unroll
      for (int i = 0; i < 8; ++i) acc[m][n][i] = 0.f;

  const int arow  = tid >> 1;
  const int ahalf = (tid & 1) * 16;
  const bf16_t* aSrc = H1 + ((size_t)e * CAP + rb * 128 + arow) * FFN + ahalf;
  const int aDstOff = arow * 32 + ahalf;

  const float* wBase = w2 + (size_t)e * FFN * DIM + (size_t)cb * 128;

  uint4  aReg[2];
  float4 bReg[4];

  {
    const uint4* s = (const uint4*)(aSrc);
    aReg[0] = s[0]; aReg[1] = s[1];
    #pragma unroll
    for (int i = 0; i < 4; ++i) {
      int fi = tid + i * 256, r = fi >> 5, c4 = (fi & 31) * 4;
      bReg[i] = *(const float4*)(wBase + (size_t)r * DIM + c4);
    }
    uint4* d = (uint4*)(Alds[0] + aDstOff);
    d[0] = aReg[0]; d[1] = aReg[1];
    #pragma unroll
    for (int i = 0; i < 4; ++i) {
      int fi = tid + i * 256, r = fi >> 5, c4 = (fi & 31) * 4;
      bf16_t* bd = Blds[0] + r * 128 + c4;
      bd[0] = f2bf(bReg[i].x); bd[1] = f2bf(bReg[i].y);
      bd[2] = f2bf(bReg[i].z); bd[3] = f2bf(bReg[i].w);
    }
  }
  __syncthreads();

  const int nk = FFN / 32;
  for (int kk = 0; kk < nk; ++kk) {
    const int cur = kk & 1, nxt = cur ^ 1;
    if (kk + 1 < nk) {
      int k0 = (kk + 1) * 32;
      const uint4* s = (const uint4*)(aSrc + k0);
      aReg[0] = s[0]; aReg[1] = s[1];
      #pragma unroll
      for (int i = 0; i < 4; ++i) {
        int fi = tid + i * 256, r = fi >> 5, c4 = (fi & 31) * 4;
        bReg[i] = *(const float4*)(wBase + (size_t)(k0 + r) * DIM + c4);
      }
    }
    {
      v16bf bf0 = load_b_frag(Blds[cur], waveCol, lane);
      v16bf bf1 = load_b_frag(Blds[cur], waveCol + 16, lane);
      #pragma unroll
      for (int m = 0; m < 4; ++m) {
        v16bf af = load_a_frag(Alds[cur], waveRow + m * 16, lane);
        acc[m][0] = wmma_bf16(af, bf0, acc[m][0]);
        acc[m][1] = wmma_bf16(af, bf1, acc[m][1]);
      }
    }
    if (kk + 1 < nk) {
      uint4* d = (uint4*)(Alds[nxt] + aDstOff);
      d[0] = aReg[0]; d[1] = aReg[1];
      #pragma unroll
      for (int i = 0; i < 4; ++i) {
        int fi = tid + i * 256, r = fi >> 5, c4 = (fi & 31) * 4;
        bf16_t* bd = Blds[nxt] + r * 128 + c4;
        bd[0] = f2bf(bReg[i].x); bd[1] = f2bf(bReg[i].y);
        bd[2] = f2bf(bReg[i].z); bd[3] = f2bf(bReg[i].w);
      }
    }
    __syncthreads();
  }

  // epilogue: + b2 + residual feat[sort[row]] ; scatter to out[sort[row]]
  const int nLane = lane & 15;
  const int mOff  = (lane >> 4) * 8;
  #pragma unroll
  for (int m = 0; m < 4; ++m)
    #pragma unroll
    for (int n = 0; n < 2; ++n) {
      int col = cb * 128 + waveCol + n * 16 + nLane;
      float bias = b2[e * DIM + col];
      #pragma unroll
      for (int i = 0; i < 8; ++i) {
        int row = rb * 128 + waveRow + m * 16 + mOff + i;
        int srcRow = sortIdx[e * CAP + row];
        float res = x[(size_t)srcRow * DIM + col];
        out[(size_t)srcRow * DIM + col] = acc[m][n][i] + bias + res;
      }
    }
}

// ---------------------------------------------------------------------------
// launcher
// ---------------------------------------------------------------------------
extern "C" void kernel_launch(void* const* d_in, const int* in_sizes, int n_in,
                              void* d_out, int out_size, void* d_ws, size_t ws_size,
                              hipStream_t stream) {
  (void)in_sizes; (void)n_in; (void)out_size; (void)ws_size;
  const float* x    = (const float*)d_in[0];
  const float* cent = (const float*)d_in[1];
  const float* ln_g = (const float*)d_in[2];
  const float* ln_b = (const float*)d_in[3];
  const float* w1   = (const float*)d_in[4];
  const float* b1   = (const float*)d_in[5];
  const float* w2   = (const float*)d_in[6];
  const float* b2   = (const float*)d_in[7];
  float* out = (float*)d_out;

  // workspace layout (256B aligned)
  char* ws = (char*)d_ws;
  size_t off = 0;
  bf16_t* norm = (bf16_t*)(ws + off);        off += (size_t)T_TOK * DIM * 2;      // 16 MB
  float* scores = (float*)(ws + off);        off += (size_t)T_TOK * NEXP * 4;     // 256 KB
  int* sortIdx = (int*)(ws + off);           off += (size_t)T_TOK * 4;            // 32 KB
  unsigned int* taken = (unsigned int*)(ws + off); off += 1024;                   // 1 KB
  off = (off + 255) & ~(size_t)255;
  bf16_t* H1 = (bf16_t*)(ws + off);          off += (size_t)NEXP * CAP * FFN * 2; // 64 MB

  ln_scores_kernel<<<T_TOK / 8, 256, 0, stream>>>(x, cent, ln_g, ln_b, norm, scores);
  assign_kernel<<<1, 1024, 0, stream>>>(scores, sortIdx, taken);
  gemm1_kernel<<<dim3(FFN / 128, CAP / 128, NEXP), 256, 0, stream>>>(norm, w1, b1, sortIdx, H1);
  gemm2_kernel<<<dim3(DIM / 128, CAP / 128, NEXP), 256, 0, stream>>>(H1, w2, b2, sortIdx, x, out);
}
